// Wrapped8BitLinear_47218870452646
// MI455X (gfx1250) — matrix operand order repaired and checked by reading, not compiled, not run on hardware
//
#include <hip/hip_runtime.h>
#include <hip/hip_bf16.h>

typedef __attribute__((ext_vector_type(16))) _Float16 v16h;
typedef __attribute__((ext_vector_type(8)))  _Float16 v8h;
typedef __attribute__((ext_vector_type(8)))  float    v8f;
typedef __attribute__((ext_vector_type(4)))  int      v4i;

#define OUT_N 16384
#define IN_K  4096
#define TOKENS 32

// One wave computes a 32(M) x 16(N) output tile, streaming K=4096 in steps of 32
// through v_wmma_f32_16x16x32_f16. 8 waves / block, 128 blocks -> 1024 N-tiles.
//
// Key points:
//  - int8 codes are exact in fp16, so the inner loop feeds raw codes to WMMA and
//    the per-row scale (SCB/127) is folded into the fp32 epilogue (saves 16
//    v_mul_f16 per lane per K-step vs scaling each weight).
//  - CB (268 MB, read once) uses non-temporal loads + a prefetch stream 8
//    K-steps ahead (matches the 128 B/row/iter consumption rate).
__global__ __launch_bounds__(256) void
Wrapped8BitLinear_47218870452646_kernel(const int* __restrict__ CB,        // [OUT_N, IN_K] int8 codes in int32
                                        const _Float16* __restrict__ SCB,  // [OUT_N]
                                        const _Float16* __restrict__ bias, // [OUT_N]
                                        const _Float16* __restrict__ X,    // [TOKENS, IN_K]
                                        float* __restrict__ out)           // [TOKENS, OUT_N]
{
    const int lane  = threadIdx.x & 31;
    const int wave  = threadIdx.x >> 5;
    const int ntile = blockIdx.x * 8 + wave;     // 0..1023
    const int n0    = ntile * 16;

    const int col = lane & 15;                   // N column within tile == CB row offset
    const int hi  = lane >> 4;                   // half-wave select
    const int row = n0 + col;                    // CB / SCB / bias row this lane owns

    // Reference computes the scale in fp16; keep that quantization, then use fp32.
    const float scale = (float)(_Float16)(SCB[row] / (_Float16)127);

    // WMMA 16-bit operand K offsets per half-wave:
    //  B (32x16): lanes 0-15 hold K+0..15, lanes 16-31 hold K+16..31 (contiguous)
    //  A (16x32): lanes 0-15 hold K+0..15, lanes 16-31 hold K+16..31 (contiguous)
    const int koffB = hi * 16;
    const int koffA = hi * 16;

    const int* __restrict__ cbp = CB + (long)row * IN_K;
    const _Float16* __restrict__ x0 = X + (long)col * IN_K;          // A tile 0: tokens 0..15
    const _Float16* __restrict__ x1 = X + (long)(col + 16) * IN_K;   // A tile 1: tokens 16..31

    v8f acc0 = {};
    v8f acc1 = {};

#pragma unroll 2
    for (int k = 0; k < IN_K; k += 32) {
        // keep the HBM weight stream ahead (global_prefetch_b8)
        __builtin_prefetch(cbp + k + koffB + 256, 0, 1);

        // ---- B tile: 16 int32 codes -> fp16 (exact; scale folded into epilogue) ----
        const v4i* __restrict__ bs = (const v4i*)(cbp + k + koffB);
        const v4i q0 = __builtin_nontemporal_load(bs + 0);
        const v4i q1 = __builtin_nontemporal_load(bs + 1);
        const v4i q2 = __builtin_nontemporal_load(bs + 2);
        const v4i q3 = __builtin_nontemporal_load(bs + 3);
        v16h B;
#pragma unroll
        for (int i = 0; i < 4; ++i) {
            B[i]      = (_Float16)q0[i];
            B[4 + i]  = (_Float16)q1[i];
            B[8 + i]  = (_Float16)q2[i];
            B[12 + i] = (_Float16)q3[i];
        }

        // ---- A tiles: two b128 loads each (x stays hot in L2/WGP$) ----
        const v8h a0lo = *(const v8h*)(x0 + k + koffA);
        const v8h a0hi = *(const v8h*)(x0 + k + koffA + 8);
        const v8h a1lo = *(const v8h*)(x1 + k + koffA);
        const v8h a1hi = *(const v8h*)(x1 + k + koffA + 8);
        const v16h A0 = __builtin_shufflevector(a0lo, a0hi, 0,1,2,3,4,5,6,7,8,9,10,11,12,13,14,15);
        const v16h A1 = __builtin_shufflevector(a1lo, a1hi, 0,1,2,3,4,5,6,7,8,9,10,11,12,13,14,15);

        acc0 = __builtin_amdgcn_wmma_f32_16x16x32_f16(false, A0, false, B,
                                                      (short)0, acc0, false, false);
        acc1 = __builtin_amdgcn_wmma_f32_16x16x32_f16(false, A1, false, B,
                                                      (short)0, acc1, false, false);
    }

    // ---- epilogue: acc*scale + bias, round through fp16 (reference output dtype) ----
    const float bv = (float)bias[row];
#pragma unroll
    for (int r = 0; r < 8; ++r) {
        const int m = r + hi * 8;                 // D layout: VGPR r -> M = r + 8*hi, N = col
        const float y0 = (float)(_Float16)(acc0[r] * scale + bv);
        const float y1 = (float)(_Float16)(acc1[r] * scale + bv);
        __builtin_nontemporal_store(y0, out + (long)m * OUT_N + row);
        __builtin_nontemporal_store(y1, out + (long)(m + 16) * OUT_N + row);
    }
}

extern "C" void kernel_launch(void* const* d_in, const int* in_sizes, int n_in,
                              void* d_out, int out_size, void* d_ws, size_t ws_size,
                              hipStream_t stream) {
    const int*      CB   = (const int*)d_in[0];
    const _Float16* SCB  = (const _Float16*)d_in[1];
    const _Float16* bias = (const _Float16*)d_in[2];
    const _Float16* x    = (const _Float16*)d_in[3];
    float*          out  = (float*)d_out;

    // 1024 N-tiles, 8 waves per 256-thread block -> 128 blocks
    dim3 grid(OUT_N / 16 / 8);
    dim3 block(256);
    Wrapped8BitLinear_47218870452646_kernel<<<grid, block, 0, stream>>>(CB, SCB, bias, x, out);
}
